// GATLayer_24575802867719
// MI455X (gfx1250) — matrix operand order repaired and checked by reading, not compiled, not run on hardware
//
#include <hip/hip_runtime.h>
#include <hip/hip_bf16.h>
#include <math.h>

// ---------------------------------------------------------------------------
// GAT layer for MI455X (gfx1250): bf16 WMMA GEMMs + fp32 softmax/aggregate
// bf16 packing done with one v_perm_b32 per pair (truncation).
// ---------------------------------------------------------------------------

typedef __attribute__((ext_vector_type(16))) __bf16        v16bf;
typedef __attribute__((ext_vector_type(8)))  float         v8f;
typedef __attribute__((ext_vector_type(8)))  unsigned int  v8u;

union BfPack { v8u u; v16bf b; };

#define IN_N  256   // node in-dim
#define OUT_N 64    // node out-dim
#define CAT   192   // edge concat dim
#define OUT_E 64    // edge out-dim

// pack two f32 into a bf16 pair (truncate): one v_perm_b32
// D.byte0 = lo.byte2, D.byte1 = lo.byte3, D.byte2 = hi.byte2, D.byte3 = hi.byte3
__device__ __forceinline__ unsigned bf2(float lo, float hi) {
    return __builtin_amdgcn_perm(__float_as_uint(hi), __float_as_uint(lo), 0x07060302u);
}

// monotone float -> uint key (total order) for atomicMax-based segment max
__device__ __forceinline__ unsigned fkey(float x) {
    unsigned u = __float_as_uint(x);
    return (u & 0x80000000u) ? ~u : (u | 0x80000000u);
}
__device__ __forceinline__ float funkey(unsigned k) {
    unsigned u = (k & 0x80000000u) ? (k ^ 0x80000000u) : ~k;
    return __uint_as_float(u);
}

// load 8 contiguous f32 starting at p (32B aligned) into packed bf16 pairs u[o..o+3]
__device__ __forceinline__ void load8_pack(const float* __restrict__ p, BfPack& A, int o) {
    float4 f0 = *(const float4*)(p);
    float4 f1 = *(const float4*)(p + 4);
    A.u[o + 0] = bf2(f0.x, f0.y);
    A.u[o + 1] = bf2(f0.z, f0.w);
    A.u[o + 2] = bf2(f1.x, f1.y);
    A.u[o + 3] = bf2(f1.z, f1.w);
}

// ---------------------------------------------------------------------------
// Kernel 0: zero h (in d_out), s, maxkey
// ---------------------------------------------------------------------------
__global__ void gat_init_kernel(float* __restrict__ h, float* __restrict__ s,
                                unsigned* __restrict__ mk, int Nh, int Nn) {
    int i = blockIdx.x * blockDim.x + threadIdx.x;
    if (i < Nh) h[i] = 0.0f;
    if (i < Nn) { s[i] = 0.0f; mk[i] = 0u; }
}

// ---------------------------------------------------------------------------
// Kernel 1: z = nfeats @ W_fc.T   [N,256] x [64,256]^T -> [N,64]
// One wave = 16x16 output tile, K loop of 8x (k=32). Block = 32 rows x 64 cols.
// ---------------------------------------------------------------------------
__global__ __launch_bounds__(256)
void gat_node_proj_kernel(const float* __restrict__ nf, const float* __restrict__ Wfc,
                          float* __restrict__ z, int N) {
    const int lane = threadIdx.x & 31;
    const int w    = threadIdx.x >> 5;
    const int m    = lane & 15;
    const int hi   = lane >> 4;
    const int rowBase = blockIdx.x * 32 + (w >> 2) * 16;
    const int n0      = (w & 3) * 16;

    int rA = rowBase + m;
    if (rA >= N) rA = N - 1;
    const float* pa_row = nf  + (size_t)rA * IN_N + hi * 8;         // A: K-half per lane group
    const float* pb_row = Wfc + (size_t)(n0 + m) * IN_N + hi * 8;   // B: W_fc[n, k]

    v8f c = {};
#pragma unroll
    for (int k0 = 0; k0 < IN_N; k0 += 32) {
        const float* pa = pa_row + k0;
        const float* pb = pb_row + k0;
        if (k0 + 32 < IN_N) {
            __builtin_prefetch(pa + 32, 0, 3);   // WGP-scope: next K-tile, same wave
            __builtin_prefetch(pb + 32, 0, 3);
        }
        BfPack A, B;
        load8_pack(pa,      A, 0);   // K = base .. base+7
        load8_pack(pa + 16, A, 4);   // K = base+16 .. base+23
        load8_pack(pb,      B, 0);   // K = base .. base+7
        load8_pack(pb + 16, B, 4);   // K = base+16 .. base+23
        c = __builtin_amdgcn_wmma_f32_16x16x32_bf16(false, A.b, false, B.b,
                                                    (short)0, c, false, false);
    }

#pragma unroll
    for (int r = 0; r < 8; ++r) {
        int row = rowBase + r + hi * 8;
        if (row < N) z[(size_t)row * OUT_N + n0 + m] = c[r];
    }
}

// ---------------------------------------------------------------------------
// Kernel 2: edge MLP.  stacked = [z[src] | efeat | z[dst]] (192)
//   feat_e = leaky_relu(stacked @ W_e.T + b_e)   -> d_out (feat part)
//   a[e]   = feat_e . w_c                        -> ws (via LDS reduce)
//   maxkey[dst[e]] = max(..., key(a[e]))         (segment max)
// ---------------------------------------------------------------------------
__global__ __launch_bounds__(256)
void gat_edge_mlp_kernel(const float* __restrict__ z, const float* __restrict__ ef,
                         const int* __restrict__ src, const int* __restrict__ dst,
                         const float* __restrict__ We, const float* __restrict__ be,
                         const float* __restrict__ wc,
                         float* __restrict__ feat_out, float* __restrict__ a_ws,
                         unsigned* __restrict__ maxkey, int E) {
    __shared__ float lds_a[32];
    if (threadIdx.x < 32) lds_a[threadIdx.x] = 0.0f;
    __syncthreads();

    const int lane = threadIdx.x & 31;
    const int w    = threadIdx.x >> 5;
    const int m    = lane & 15;
    const int hi   = lane >> 4;
    const int eBase = blockIdx.x * 32 + (w >> 2) * 16;
    const int n0    = (w & 3) * 16;

    int eA = eBase + m;
    if (eA >= E) eA = E - 1;
    const int se = src[eA];
    const int de = dst[eA];
    // concat segments: [0,64) = z[src], [64,128) = efeat, [128,192) = z[dst]
    const float* seg0 = z  + (size_t)se * OUT_N;
    const float* seg1 = ef + (size_t)eA * 64;
    const float* seg2 = z  + (size_t)de * OUT_N;

    const float* pb_row = We + (size_t)(n0 + m) * CAT + hi * 8;

    v8f c = {};
#pragma unroll
    for (int k0 = 0; k0 < CAT; k0 += 32) {
        const int seg = k0 >> 6;
        const float* base = (seg == 0) ? seg0 : (seg == 1) ? seg1 : seg2;
        const float* pa = base + (k0 & 63) + hi * 8;
        const float* pb = pb_row + k0;
        if (k0 + 32 < CAT) __builtin_prefetch(pb + 32, 0, 3);
        BfPack A, B;
        load8_pack(pa,      A, 0);
        load8_pack(pa + 16, A, 4);
        load8_pack(pb,      B, 0);
        load8_pack(pb + 16, B, 4);
        c = __builtin_amdgcn_wmma_f32_16x16x32_bf16(false, A.b, false, B.b,
                                                    (short)0, c, false, false);
    }

    const float bn  = be[n0 + m];
    const float wcn = wc[n0 + m];
#pragma unroll
    for (int r = 0; r < 8; ++r) {
        int e = eBase + r + hi * 8;
        float v = c[r] + bn;
        v = (v > 0.0f) ? v : 0.01f * v;           // leaky_relu, slope 0.01
        if (e < E) {
            feat_out[(size_t)e * OUT_E + n0 + m] = v;
            atomicAdd(&lds_a[(w >> 2) * 16 + r + hi * 8], v * wcn);
        }
    }
    __syncthreads();

    if (threadIdx.x < 32) {
        int e = blockIdx.x * 32 + threadIdx.x;
        if (e < E) {
            float av = lds_a[threadIdx.x];
            a_ws[e] = av;
            atomicMax(&maxkey[dst[e]], fkey(av));
        }
    }
}

// ---------------------------------------------------------------------------
// Kernel 3: ex[e] = exp(a[e] - max[dst[e]]);  s[dst[e]] += ex[e]
// ---------------------------------------------------------------------------
__global__ void gat_softmax_sum_kernel(const float* __restrict__ a, const int* __restrict__ dst,
                                       const unsigned* __restrict__ maxkey,
                                       float* __restrict__ ex, float* __restrict__ s, int E) {
    int e = blockIdx.x * blockDim.x + threadIdx.x;
    if (e >= E) return;
    int d = dst[e];
    float x = expf(a[e] - funkey(maxkey[d]));
    ex[e] = x;
    atomicAdd(&s[d], x);
}

// ---------------------------------------------------------------------------
// Kernel 4: h[dst[e]] += (ex[e]/s[dst[e]]) * z[src[e]], one thread per (e, col)
// ---------------------------------------------------------------------------
__global__ void gat_aggregate_kernel(const float* __restrict__ z, const int* __restrict__ src,
                                     const int* __restrict__ dst, const float* __restrict__ ex,
                                     const float* __restrict__ s, float* __restrict__ h,
                                     long total) {
    long t = (long)blockIdx.x * blockDim.x + threadIdx.x;
    if (t >= total) return;
    int e = (int)(t >> 6);
    int j = (int)(t & 63);
    int d = dst[e];
    float alpha = ex[e] / s[d];
    atomicAdd(&h[(size_t)d * OUT_N + j], alpha * z[(size_t)src[e] * OUT_N + j]);
}

// ---------------------------------------------------------------------------
// launch
// ---------------------------------------------------------------------------
extern "C" void kernel_launch(void* const* d_in, const int* in_sizes, int n_in,
                              void* d_out, int out_size, void* d_ws, size_t ws_size,
                              hipStream_t stream) {
    const float* nfeats = (const float*)d_in[0];
    const float* efeats = (const float*)d_in[1];
    const int*   src    = (const int*)  d_in[2];
    const int*   dst    = (const int*)  d_in[3];
    const float* W_fc   = (const float*)d_in[4];
    const float* W_e    = (const float*)d_in[5];
    const float* b_e    = (const float*)d_in[6];
    const float* w_c    = (const float*)d_in[7];

    const int N = in_sizes[0] / IN_N;   // 50000
    const int E = in_sizes[2];          // 800000

    // outputs: h [N,64] then feat_e [E,64]
    float* h        = (float*)d_out;
    float* feat_out = h + (size_t)N * OUT_N;

    // workspace: z [N,64] | a [E] | ex [E] | s [N] | maxkey [N]
    float*    z  = (float*)d_ws;
    float*    a  = z  + (size_t)N * OUT_N;
    float*    ex = a  + E;
    float*    s  = ex + E;
    unsigned* mk = (unsigned*)(s + N);

    const int Nh = N * OUT_N;
    gat_init_kernel<<<(Nh + 255) / 256, 256, 0, stream>>>(h, s, mk, Nh, N);

    gat_node_proj_kernel<<<(N + 31) / 32, 256, 0, stream>>>(nfeats, W_fc, z, N);

    gat_edge_mlp_kernel<<<(E + 31) / 32, 256, 0, stream>>>(
        z, efeats, src, dst, W_e, b_e, w_c, feat_out, a, mk, E);

    gat_softmax_sum_kernel<<<(E + 255) / 256, 256, 0, stream>>>(a, dst, mk, ex, s, E);

    const long total = (long)E * OUT_N;
    gat_aggregate_kernel<<<(int)((total + 255) / 256), 256, 0, stream>>>(
        z, src, dst, ex, s, h, total);
}
